// GatedDeltaNet_25640954757687
// MI455X (gfx1250) — matrix operand order repaired and checked by reading, not compiled, not run on hardware
//
#include <hip/hip_runtime.h>

// ---------------------------------------------------------------------------
// GatedDeltaNet for MI455X (gfx1250, wave32).
//   B=2, T=2048, HID=2048, H=16, Dh=128, conv K=4.
// GEMMs: v_wmma_f32_16x16x32_f16, f16 operands pre-staged in global memory
// (x converted once, weights converted+transposed once), tiles DMA'd into
// LDS with global_load_async_to_lds_b128 (ASYNCcnt), double-buffered, with
// the last K-step peeled so the steady-state loop is branch-free.
// Block tile 128x128, 8 waves (4x2), each wave 32x64 = 2x4 WMMA tiles.
// Scan keeps the 128x128 state in registers (64 f32 per thread, 256 threads).
// ---------------------------------------------------------------------------

typedef __attribute__((ext_vector_type(16))) _Float16 v16h;
typedef __attribute__((ext_vector_type(8)))  _Float16 v8h;
typedef __attribute__((ext_vector_type(4)))  _Float16 v4h;
typedef __attribute__((ext_vector_type(8)))  float    v8f;

constexpr int BQ   = 2;
constexpr int TQ   = 2048;
constexpr int HIDQ = 2048;
constexpr int HQ   = 16;
constexpr int DHQ  = 128;
constexpr int KDQ  = 2048;          // H * Dh
constexpr int MROWS = BQ * TQ;      // 4096

// LDS row stride for f16 tiles: 40 halves (80B) -> b128 reads stay 16B aligned
// and 80*dr mod 256B != 0 for dr in 1..15 -> conflict-free across 16 lanes.
constexpr int LSTR   = 40;
constexpr int A_BUF  = 128 * LSTR;  // halves per A buffer (128 rows x 32 k)
constexpr int B_BUF  = 128 * LSTR;  // halves per B buffer (128 n-rows x 32 k)

// ---------------------------------------------------------------------------
// async DMA: 16 bytes global -> LDS per lane, tracked by ASYNCcnt.
// ---------------------------------------------------------------------------
__device__ __forceinline__ void async_b128(_Float16* lds_ptr, const _Float16* gptr) {
  const unsigned lds_off = (unsigned)(size_t)lds_ptr;   // generic->LDS = trunc
  asm volatile("global_load_async_to_lds_b128 %0, %1, off"
               :: "v"(lds_off), "v"(gptr) : "memory");
}
__device__ __forceinline__ void wait_async0() {
  asm volatile("s_wait_asynccnt 0x0" ::: "memory");
}

// ---------------------------------------------------------------------------
// WMMA GEMM: C[M,N] = A[M,K] * Bt[N,K]^T.  A,Bt f16 row-major, C f32.
// ---------------------------------------------------------------------------
__global__ void __launch_bounds__(256)
wmma_gemm_f16(const _Float16* __restrict__ A, const _Float16* __restrict__ Bt,
              float* __restrict__ C, int Ndim, int Kdim) {
  __shared__ __align__(16) _Float16 Alds[2 * A_BUF];
  __shared__ __align__(16) _Float16 Blds[2 * B_BUF];

  const int tid  = threadIdx.x;
  const int lane = tid & 31;
  const int wid  = tid >> 5;
  const int wm   = wid & 3;      // wave row 0..3  (32 M-rows each)
  const int wn   = wid >> 2;     // wave col 0..1  (64 N-cols each)
  const int r    = lane & 15;
  const int hsel = lane >> 4;    // half-wave select
  const int bm   = blockIdx.y * 128;
  const int bn   = blockIdx.x * 128;

  // per-thread copy chunks (16B = 8 halves each); A: 2 chunks, B: 2 chunks
  const int rA0 = tid >> 2,          cA0 = (tid & 3) << 3;
  const int rA1 = (tid + 256) >> 2,  cA1 = ((tid + 256) & 3) << 3;
  const _Float16* gA0 = A  + (size_t)(bm + rA0) * Kdim + cA0;
  const _Float16* gA1 = A  + (size_t)(bm + rA1) * Kdim + cA1;
  const _Float16* gB0 = Bt + (size_t)(bn + rA0) * Kdim + cA0;
  const _Float16* gB1 = Bt + (size_t)(bn + rA1) * Kdim + cA1;
  _Float16* lA0 = Alds + rA0 * LSTR + cA0;
  _Float16* lA1 = Alds + rA1 * LSTR + cA1;
  _Float16* lB0 = Blds + rA0 * LSTR + cA0;
  _Float16* lB1 = Blds + rA1 * LSTR + cA1;

  v8f acc[2][4] = {};
  const int nk = Kdim >> 5;

  // prologue: DMA tile 0 into buffer 0
  async_b128(lA0, gA0);
  async_b128(lA1, gA1);
  async_b128(lB0, gB0);
  async_b128(lB1, gB1);

  int cur = 0;
  for (int kt = 0; kt < nk - 1; ++kt) {        // steady state: branch-free body
    wait_async0();                 // this wave's DMA for tile kt has landed
    __syncthreads();               // everyone's has; prev buffer fully read

    const int koff = (kt + 1) << 5;            // DMA tile kt+1, other buffer
    const int nb = cur ^ 1;
    async_b128(lA0 + nb * A_BUF, gA0 + koff);
    async_b128(lA1 + nb * A_BUF, gA1 + koff);
    async_b128(lB0 + nb * B_BUF, gB0 + koff);
    async_b128(lB1 + nb * B_BUF, gB1 + koff);

    const _Float16* Ac = Alds + cur * A_BUF;
    const _Float16* Bc = Blds + cur * B_BUF;

    // A fragment: lane(r,half), element j -> k = 16*half + j (contiguous)
    v16h afrag[2], bfrag[4];
#pragma unroll
    for (int i = 0; i < 2; ++i) {
      const int base = (wm * 32 + i * 16 + r) * LSTR + hsel * 16;
      v8h lo = *(const v8h*)(&Ac[base]);
      v8h hi = *(const v8h*)(&Ac[base + 8]);
      afrag[i] = __builtin_shufflevector(lo, hi, 0, 1, 2, 3, 4, 5, 6, 7,
                                         8, 9, 10, 11, 12, 13, 14, 15);
    }
    // B fragment: lane(c=r,half), element j -> K = 16*half + j (contiguous)
#pragma unroll
    for (int j = 0; j < 4; ++j) {
      const int base = (wn * 64 + j * 16 + r) * LSTR + hsel * 16;
      v8h lo = *(const v8h*)(&Bc[base]);
      v8h hi = *(const v8h*)(&Bc[base + 8]);
      bfrag[j] = __builtin_shufflevector(lo, hi, 0, 1, 2, 3, 4, 5, 6, 7,
                                         8, 9, 10, 11, 12, 13, 14, 15);
    }
#pragma unroll
    for (int i = 0; i < 2; ++i)
#pragma unroll
      for (int j = 0; j < 4; ++j)
        acc[i][j] = __builtin_amdgcn_wmma_f32_16x16x32_f16(
            false, afrag[i], false, bfrag[j], (short)0, acc[i][j], false, false);
    cur ^= 1;
  }

  // peeled last K-step (no further DMA)
  wait_async0();
  __syncthreads();
  {
    const _Float16* Ac = Alds + cur * A_BUF;
    const _Float16* Bc = Blds + cur * B_BUF;
    v16h afrag[2], bfrag[4];
#pragma unroll
    for (int i = 0; i < 2; ++i) {
      const int base = (wm * 32 + i * 16 + r) * LSTR + hsel * 16;
      v8h lo = *(const v8h*)(&Ac[base]);
      v8h hi = *(const v8h*)(&Ac[base + 8]);
      afrag[i] = __builtin_shufflevector(lo, hi, 0, 1, 2, 3, 4, 5, 6, 7,
                                         8, 9, 10, 11, 12, 13, 14, 15);
    }
#pragma unroll
    for (int j = 0; j < 4; ++j) {
      const int base = (wn * 64 + j * 16 + r) * LSTR + hsel * 16;
      v8h lo = *(const v8h*)(&Bc[base]);
      v8h hi = *(const v8h*)(&Bc[base + 8]);
      bfrag[j] = __builtin_shufflevector(lo, hi, 0, 1, 2, 3, 4, 5, 6, 7,
                                         8, 9, 10, 11, 12, 13, 14, 15);
    }
#pragma unroll
    for (int i = 0; i < 2; ++i)
#pragma unroll
      for (int j = 0; j < 4; ++j)
        acc[i][j] = __builtin_amdgcn_wmma_f32_16x16x32_f16(
            false, afrag[i], false, bfrag[j], (short)0, acc[i][j], false, false);
  }

  // epilogue: C/D layout -> row = tile_m + 8*half + vgpr, col = tile_n + r
#pragma unroll
  for (int i = 0; i < 2; ++i) {
    const int mbase = bm + wm * 32 + i * 16 + hsel * 8;
#pragma unroll
    for (int j = 0; j < 4; ++j) {
      const int col = bn + wn * 64 + j * 16 + r;
#pragma unroll
      for (int vv = 0; vv < 8; ++vv)
        C[(size_t)(mbase + vv) * Ndim + col] = acc[i][j][vv];
    }
  }
}

// ---------------------------------------------------------------------------
// f32 -> f16 elementwise (4 per thread)
// ---------------------------------------------------------------------------
__global__ void __launch_bounds__(256)
cvt_f16_kernel(const float* __restrict__ in, _Float16* __restrict__ out) {
  const size_t i = ((size_t)blockIdx.x * blockDim.x + threadIdx.x) * 4;
  const float4 v = *(const float4*)(in + i);
  v4h h = { (_Float16)v.x, (_Float16)v.y, (_Float16)v.z, (_Float16)v.w };
  *(v4h*)(out + i) = h;
}

// ---------------------------------------------------------------------------
// f32 [R,C] -> f16 [C,R] tiled transpose-convert (32x32 tiles via LDS)
// ---------------------------------------------------------------------------
__global__ void __launch_bounds__(256)
transpose_cvt_kernel(const float* __restrict__ in, _Float16* __restrict__ out,
                     int R, int Ccols) {
  __shared__ float tile[32][33];
  const int c0 = blockIdx.x * 32, r0 = blockIdx.y * 32;
  const int tx = threadIdx.x & 31, ty = threadIdx.x >> 5;   // 8 rows / pass
#pragma unroll
  for (int i = 0; i < 32; i += 8)
    tile[ty + i][tx] = in[(size_t)(r0 + ty + i) * Ccols + c0 + tx];
  __syncthreads();
#pragma unroll
  for (int i = 0; i < 32; i += 8)
    out[(size_t)(c0 + ty + i) * R + r0 + tx] = (_Float16)tile[tx][ty + i];
}

// ---------------------------------------------------------------------------
// beta = sigmoid(x@Wb + bb) ; alpha = exp(-exp(A_log) * softplus(x@Wgk+bgk+dt))
// ---------------------------------------------------------------------------
__global__ void __launch_bounds__(256)
gates_kernel(const float* __restrict__ x, const float* __restrict__ Wb,
             const float* __restrict__ bb, const float* __restrict__ Wgk,
             const float* __restrict__ bgk, const float* __restrict__ A_log,
             const float* __restrict__ dt_bias,
             float* __restrict__ beta, float* __restrict__ alpha) {
  const int idx = blockIdx.x * blockDim.x + threadIdx.x;   // row*H + h
  const int h   = idx & (HQ - 1);
  const int row = idx >> 4;
  const float* xr = x + (size_t)row * HIDQ;
  float sb = 0.f, sg = 0.f;
  for (int kk = 0; kk < HIDQ; ++kk) {
    const float xv = xr[kk];
    sb = fmaf(xv, Wb[kk * HQ + h], sb);
    sg = fmaf(xv, Wgk[kk * HQ + h], sg);
  }
  beta[idx] = 1.f / (1.f + expf(-(sb + bb[h])));
  const float z  = sg + bgk[h] + dt_bias[h];
  const float sp = (z > 20.f) ? z : log1pf(expf(z));
  alpha[idx] = expf(-expf(A_log[h]) * sp);
}

// ---------------------------------------------------------------------------
// Depthwise causal conv (K=4) + bias + SiLU. Coalesced over channels.
// ---------------------------------------------------------------------------
__global__ void __launch_bounds__(256)
conv_silu_kernel(const float* __restrict__ in, const float* __restrict__ w,
                 const float* __restrict__ bias, float* __restrict__ out) {
  const size_t idx = (size_t)blockIdx.x * blockDim.x + threadIdx.x;
  const int c = (int)(idx % KDQ);
  const int t = (int)((idx / KDQ) % TQ);
  const int b = (int)(idx / ((size_t)KDQ * TQ));
  float acc = bias[c];
#pragma unroll
  for (int j = 0; j < 4; ++j) {
    const int ts = t - 3 + j;
    if (ts >= 0) acc = fmaf(in[((size_t)b * TQ + ts) * KDQ + c], w[c * 4 + j], acc);
  }
  out[idx] = acc / (1.f + expf(-acc));   // SiLU
}

// ---------------------------------------------------------------------------
// RoPE (reference pairing: freq index j=(2i) mod 64,
// out[i]=x[2i]*c-x[2i+1]*s ; out[64+i]=x[2i]*s+x[2i+1]*c) + L2 norm, in place.
// One wave per (b,t,h): all loads precede all stores in the wave.
// ---------------------------------------------------------------------------
__global__ void __launch_bounds__(256)
rope_l2n_kernel(float* __restrict__ qk) {
  const int gtid = blockIdx.x * 256 + threadIdx.x;
  const int w    = gtid >> 5;                 // (b*T+t)*H + h
  const int lane = threadIdx.x & 31;
  const int t    = (w >> 4) & (TQ - 1);
  float* p = qk + (size_t)w * DHQ;
  const float LOG_BASE = 9.210340371976184f;  // ln(10000)
  float out[4];
  float sumsq = 0.f;
#pragma unroll
  for (int u = 0; u < 2; ++u) {
    const int i = lane + u * 32;
    const int j = (2 * i) & 63;
    const float inv = expf(-(float)j * (LOG_BASE / 64.f));
    const float ang = (float)t * inv;
    const float c = cosf(ang), s = sinf(ang);
    const float x1 = p[2 * i], x2 = p[2 * i + 1];
    out[u * 2 + 0] = x1 * c - x2 * s;
    out[u * 2 + 1] = x1 * s + x2 * c;
    sumsq += out[u * 2] * out[u * 2] + out[u * 2 + 1] * out[u * 2 + 1];
  }
#pragma unroll
  for (int off = 16; off > 0; off >>= 1) sumsq += __shfl_xor(sumsq, off, 32);
  const float sc = 1.f / fmaxf(sqrtf(sumsq), 1e-12f);
#pragma unroll
  for (int u = 0; u < 2; ++u) {
    const int i = lane + u * 32;
    p[i]      = out[u * 2 + 0] * sc;
    p[i + 64] = out[u * 2 + 1] * sc;
  }
}

// ---------------------------------------------------------------------------
// Sequential delta-rule scan. One block per (b,h); S[128][128] in registers,
// thread (ti,tj) owns d in [8ti,8ti+8), e in [8tj,8tj+8).
//   o_e  = sum_d q_d S_de + Dp*(q.k)*v_e          (old S)
//   Sk_d = sum_e S_de k_e
//   S_de <- a*(S_de - b*Sk_d*k_e) + b*v_d*k_e
// ---------------------------------------------------------------------------
__global__ void __launch_bounds__(256)
scan_kernel(const float* __restrict__ q, const float* __restrict__ k,
            const float* __restrict__ v, const float* __restrict__ alpha,
            const float* __restrict__ beta, const float* __restrict__ Dp,
            float* __restrict__ o) {
  const int bh = blockIdx.x;          // b*H + h
  const int b  = bh >> 4;
  const int h  = bh & 15;
  const int tid = threadIdx.x;
  const int ti = tid & 15, tj = tid >> 4;
  const int d0 = ti * 8, e0 = tj * 8;
  const float dp = Dp[h];

  __shared__ float q_sh[128], k_sh[128], v_sh[128], sk_sh[128];
  __shared__ float o_red[16][128];
  __shared__ float s_red[16][128];
  __shared__ float qk4[4];

  float S[64];
#pragma unroll
  for (int i = 0; i < 64; ++i) S[i] = 0.f;

  for (int t = 0; t < TQ; ++t) {
    const size_t base = ((size_t)(b * TQ + t) * HQ + h) * DHQ;
    __syncthreads();                              // protect LDS from prev update
    if (tid < 128) {
      q_sh[tid] = q[base + tid];
      k_sh[tid] = k[base + tid];
      v_sh[tid] = v[base + tid];
    }
    __syncthreads();

    float op[8], sp[8];
#pragma unroll
    for (int ee = 0; ee < 8; ++ee) op[ee] = 0.f;
#pragma unroll
    for (int dd = 0; dd < 8; ++dd) sp[dd] = 0.f;
#pragma unroll
    for (int dd = 0; dd < 8; ++dd) {
      const float qd = q_sh[d0 + dd];
#pragma unroll
      for (int ee = 0; ee < 8; ++ee) {
        const float s = S[dd * 8 + ee];
        op[ee] = fmaf(qd, s, op[ee]);
        sp[dd] = fmaf(s, k_sh[e0 + ee], sp[dd]);
      }
    }
#pragma unroll
    for (int ee = 0; ee < 8; ++ee) o_red[ti][e0 + ee] = op[ee];
#pragma unroll
    for (int dd = 0; dd < 8; ++dd) s_red[tj][d0 + dd] = sp[dd];
    if (tid < 128) {
      float p = q_sh[tid] * k_sh[tid];
#pragma unroll
      for (int off = 16; off > 0; off >>= 1) p += __shfl_xor(p, off, 32);
      if ((tid & 31) == 0) qk4[tid >> 5] = p;
    }
    __syncthreads();
    if (tid < 128) {
      float acc_o = 0.f, acc_s = 0.f;
#pragma unroll
      for (int i = 0; i < 16; ++i) { acc_o += o_red[i][tid]; acc_s += s_red[i][tid]; }
      sk_sh[tid] = acc_s;
      const float qk = qk4[0] + qk4[1] + qk4[2] + qk4[3];
      o[base + tid] = acc_o + dp * qk * v_sh[tid];
    }
    __syncthreads();
    const float a  = alpha[(size_t)(b * TQ + t) * HQ + h];
    const float bt = beta[(size_t)(b * TQ + t) * HQ + h];
#pragma unroll
    for (int dd = 0; dd < 8; ++dd) {
      const float vD  = v_sh[d0 + dd];
      const float skD = sk_sh[d0 + dd];
#pragma unroll
      for (int ee = 0; ee < 8; ++ee) {
        const float kE = k_sh[e0 + ee];
        S[dd * 8 + ee] = a * (S[dd * 8 + ee] - bt * skD * kE) + bt * vD * kE;
      }
    }
  }
}

// ---------------------------------------------------------------------------
// og(f16) = g * silu(o * rsqrt(mean(o^2)+1e-6) * onorm_w). One wave per (b,t,h).
// ---------------------------------------------------------------------------
__global__ void __launch_bounds__(256)
gate_rms_kernel(const float* __restrict__ o, const float* __restrict__ g,
                const float* __restrict__ onw, _Float16* __restrict__ og) {
  const int gtid = blockIdx.x * 256 + threadIdx.x;
  const int w    = gtid >> 5;
  const int lane = threadIdx.x & 31;
  const size_t base = (size_t)w * DHQ + lane * 4;
  const float4 ov = *(const float4*)(o + base);
  float ss = ov.x * ov.x + ov.y * ov.y + ov.z * ov.z + ov.w * ov.w;
#pragma unroll
  for (int off = 16; off > 0; off >>= 1) ss += __shfl_xor(ss, off, 32);
  const float sc = rsqrtf(ss * (1.f / 128.f) + 1e-6f);
  const float4 gv = *(const float4*)(g + base);
  const float4 nw = *(const float4*)(onw + lane * 4);
  float r[4], oh;
  oh = ov.x * sc * nw.x; r[0] = gv.x * oh / (1.f + expf(-oh));
  oh = ov.y * sc * nw.y; r[1] = gv.y * oh / (1.f + expf(-oh));
  oh = ov.z * sc * nw.z; r[2] = gv.z * oh / (1.f + expf(-oh));
  oh = ov.w * sc * nw.w; r[3] = gv.w * oh / (1.f + expf(-oh));
  v4h hv = { (_Float16)r[0], (_Float16)r[1], (_Float16)r[2], (_Float16)r[3] };
  *(v4h*)(og + base) = hv;
}

// ---------------------------------------------------------------------------
extern "C" void kernel_launch(void* const* d_in, const int* in_sizes, int n_in,
                              void* d_out, int out_size, void* d_ws, size_t ws_size,
                              hipStream_t stream) {
  const float* x       = (const float*)d_in[0];
  const float* Wq      = (const float*)d_in[1];
  const float* Wk      = (const float*)d_in[2];
  const float* Wv      = (const float*)d_in[3];
  const float* Wg      = (const float*)d_in[4];
  const float* Wo      = (const float*)d_in[5];
  const float* Wb      = (const float*)d_in[6];
  const float* bb      = (const float*)d_in[7];
  const float* Wgk     = (const float*)d_in[8];
  const float* bgk     = (const float*)d_in[9];
  const float* cqw     = (const float*)d_in[10];
  const float* cqb     = (const float*)d_in[11];
  const float* ckw     = (const float*)d_in[12];
  const float* ckb     = (const float*)d_in[13];
  const float* cvw     = (const float*)d_in[14];
  const float* cvb     = (const float*)d_in[15];
  const float* A_log   = (const float*)d_in[16];
  const float* Dp      = (const float*)d_in[17];
  const float* dt_bias = (const float*)d_in[18];
  const float* onorm_w = (const float*)d_in[19];
  float* out = (float*)d_out;

  // ---- workspace layout --------------------------------------------------
  float* wsf = (float*)d_ws;
  const size_t SLOT = (size_t)MROWS * KDQ;     // 8,388,608 elements
  float* qp = wsf + 0 * SLOT;
  float* kp = wsf + 1 * SLOT;
  float* vp = wsf + 2 * SLOT;
  float* gp = wsf + 3 * SLOT;
  float* qc = wsf + 4 * SLOT;
  float* kc = wsf + 5 * SLOT;
  float* vc = wsf + 6 * SLOT;
  float* obuf  = qp;                 // qp dead after conv -> reuse for scan out
  float* betab  = wsf + 7 * SLOT;
  float* alphab = betab + (size_t)MROWS * HQ;
  // f16 region
  const size_t WHALF = (size_t)HIDQ * KDQ;     // 4,194,304 halves per weight
  _Float16* hbase = (_Float16*)(alphab + (size_t)MROWS * HQ);
  _Float16* xh  = hbase;                       // [M][K]
  _Float16* Wqt = xh + SLOT;                   // [N][K] transposed weights
  _Float16* Wkt = Wqt + WHALF;
  _Float16* Wvt = Wkt + WHALF;
  _Float16* Wgt = Wvt + WHALF;
  _Float16* Wot = Wgt + WHALF;
  _Float16* ogh = Wot + WHALF;                 // gated output, f16 [M][K]

  const dim3 gblk(256);
  const dim3 ggrid(KDQ / 128, MROWS / 128);    // 16 x 32 blocks
  const dim3 tgrid(KDQ / 32, HIDQ / 32);       // 64 x 64 transpose tiles

  // ---- one-time f16 prep -------------------------------------------------
  cvt_f16_kernel<<<(int)(SLOT / 1024), 256, 0, stream>>>(x, xh);
  transpose_cvt_kernel<<<tgrid, 256, 0, stream>>>(Wq, Wqt, HIDQ, KDQ);
  transpose_cvt_kernel<<<tgrid, 256, 0, stream>>>(Wk, Wkt, HIDQ, KDQ);
  transpose_cvt_kernel<<<tgrid, 256, 0, stream>>>(Wv, Wvt, HIDQ, KDQ);
  transpose_cvt_kernel<<<tgrid, 256, 0, stream>>>(Wg, Wgt, HIDQ, KDQ);
  transpose_cvt_kernel<<<tgrid, 256, 0, stream>>>(Wo, Wot, KDQ, HIDQ);

  // ---- QKVG projections (WMMA, async-LDS double buffered) ----------------
  wmma_gemm_f16<<<ggrid, gblk, 0, stream>>>(xh, Wqt, qp, KDQ, HIDQ);
  wmma_gemm_f16<<<ggrid, gblk, 0, stream>>>(xh, Wkt, kp, KDQ, HIDQ);
  wmma_gemm_f16<<<ggrid, gblk, 0, stream>>>(xh, Wvt, vp, KDQ, HIDQ);
  wmma_gemm_f16<<<ggrid, gblk, 0, stream>>>(xh, Wgt, gp, KDQ, HIDQ);

  // ---- gates (beta / alpha) ----------------------------------------------
  gates_kernel<<<(MROWS * HQ) / 256, 256, 0, stream>>>(x, Wb, bb, Wgk, bgk,
                                                       A_log, dt_bias, betab, alphab);

  // ---- causal depthwise conv + SiLU --------------------------------------
  const int nconvBlocks = (int)(((size_t)BQ * TQ * KDQ) / 256);
  conv_silu_kernel<<<nconvBlocks, 256, 0, stream>>>(qp, cqw, cqb, qc);
  conv_silu_kernel<<<nconvBlocks, 256, 0, stream>>>(kp, ckw, ckb, kc);
  conv_silu_kernel<<<nconvBlocks, 256, 0, stream>>>(vp, cvw, cvb, vc);

  // ---- RoPE + L2 normalize (in place) on q, k ----------------------------
  const int nwaveBlocks = (BQ * TQ * HQ * 32) / 256;
  rope_l2n_kernel<<<nwaveBlocks, 256, 0, stream>>>(qc);
  rope_l2n_kernel<<<nwaveBlocks, 256, 0, stream>>>(kc);

  // ---- sequential delta-rule scan, one block per (b,h) -------------------
  scan_kernel<<<BQ * HQ, 256, 0, stream>>>(qc, kc, vc, alphab, betab, Dp, obuf);

  // ---- RMS-norm gate -> f16 ----------------------------------------------
  gate_rms_kernel<<<nwaveBlocks, 256, 0, stream>>>(obuf, gp, onorm_w, ogh);

  // ---- output projection (WMMA) ------------------------------------------
  wmma_gemm_f16<<<dim3(HIDQ / 128, MROWS / 128), gblk, 0, stream>>>(ogh, Wot, out,
                                                                    HIDQ, KDQ);
}
